// UpsampleTransformerAlibi_60825326845989
// MI455X (gfx1250) — matrix operand order repaired and checked by reading, not compiled, not run on hardware
//
#include <hip/hip_runtime.h>
#include <hip/hip_bf16.h>
#include <math.h>

typedef _Float16 h16;
typedef __attribute__((ext_vector_type(16))) _Float16 v16h;
typedef __attribute__((ext_vector_type(8)))  _Float16 v8h;
typedef __attribute__((ext_vector_type(8)))  float    v8f;

// ---------------- WMMA fragment helpers (gfx1250 wave32, f32 = f16 x f16 + f32) ----
// A (16 x 32): lane m = lane&15 is the matrix row, g = lane>>4 picks K-halves.
//   vector elem e<8  -> K = k0 + g*8 + e
//   vector elem e>=8 -> K = k0 + 16 + g*8 + (e-8)
__device__ __forceinline__ v16h afrag(const h16* rowk0, int g) {
  const v8h lo = *(const v8h*)(rowk0 + g * 8);
  const v8h hi = *(const v8h*)(rowk0 + 16 + g * 8);
  v16h a;
#pragma unroll
  for (int e = 0; e < 8; ++e) { a[e] = lo[e]; a[8 + e] = hi[e]; }
  return a;
}
// B (32 x 16): lane n = lane&15 is the matrix column, elems follow the same K order as A.
__device__ __forceinline__ v16h bfrag(const h16* colk0, int g) {
  const v8h lo = *(const v8h*)(colk0 + g * 8);
  const v8h hi = *(const v8h*)(colk0 + 16 + g * 8);
  v16h b;
#pragma unroll
  for (int e = 0; e < 8; ++e) { b[e] = lo[e]; b[8 + e] = hi[e]; }
  return b;
}
__device__ __forceinline__ v8f wmma16(v16h a, v16h b, v8f c) {
  return __builtin_amdgcn_wmma_f32_16x16x32_f16(false, a, false, b, (short)0, c,
                                                false, false);
}
// C/D: element r of v8f  <->  (M = r + 8*(lane>>4), N = lane&15)

// ---------------- weight packing (f32 -> f16) --------------------------------------
// conv weights: w[co][ci][ky][kx] -> wPk[co][(ky*KS+kx)*CI + ci]
__global__ void pack_conv(const float* __restrict__ w, h16* __restrict__ dst,
                          int CO, int CI, int KS) {
  int i = blockIdx.x * 256 + threadIdx.x;
  int tot = CO * CI * KS * KS;
  if (i >= tot) return;
  int ci = i % CI;
  int rem = i / CI;
  int tap = rem % (KS * KS);
  int co = rem / (KS * KS);
  int ky = tap / KS, kx = tap % KS;
  dst[i] = (h16)w[((co * CI + ci) * KS + ky) * KS + kx];
}

__global__ void pack_mat(const float* __restrict__ w, h16* __restrict__ dst, int n) {
  int i = blockIdx.x * 256 + threadIdx.x;
  if (i < n) dst[i] = (h16)w[i];
}

// ---------------- implicit-GEMM conv (im2col tile in LDS, WMMA mainloop) -----------
// Computes out[co][y][x] = bias + sum_{ci,dy,dx} w * inL[ci][y+dy-PAD][x+dx-PAD]
// inL is `in` (Hin x Win) or, if UPS, nearest-2x upsample of it (2Hin x 2Win).
// Output dims = logical input dims. Optional residual add and ReLU.
template <int KS, int CIN, int COUT, bool UPS>
__global__ void __launch_bounds__(256)
conv_wmma(const float* __restrict__ in, int Hin, int Win,
          const h16* __restrict__ wPk, const float* __restrict__ bias,
          const float* __restrict__ addSrc, int reluOut, float* __restrict__ out) {
  constexpr int PAD = KS / 2;
  constexpr int COLS = 16 + KS - 1;
  constexpr int KTOT = KS * KS * CIN;
  constexpr int CPT = CIN / 32;  // K-chunks per tap
  __shared__ h16 smem[KS][COLS][CIN];
  const int HL = UPS ? 2 * Hin : Hin;
  const int WL = UPS ? 2 * Win : Win;
  const int x0 = blockIdx.x * 16;
  const int y = blockIdx.y;
  const int tid = threadIdx.x;

  for (int i = tid; i < KS * COLS * CIN; i += 256) {
    int ci = i % CIN;
    int rc = i / CIN;
    int col = rc % COLS;
    int row = rc / COLS;
    int gy = y - PAD + row, gx = x0 - PAD + col;
    float v = 0.f;
    if ((unsigned)gy < (unsigned)HL && (unsigned)gx < (unsigned)WL) {
      int sy = UPS ? (gy >> 1) : gy;
      int sx = UPS ? (gx >> 1) : gx;
      v = in[((size_t)ci * Hin + sy) * Win + sx];
    }
    smem[row][col][ci] = (h16)v;
  }
  __syncthreads();

  const int w = tid >> 5, lane = tid & 31, n = lane & 15, g = lane >> 4;
  const int co0 = w * 16;
  if (co0 >= COUT) return;

  v8f acc = {};
  const h16* wrow = wPk + (size_t)(co0 + n) * KTOT;  // A row m = lane&15
#pragma unroll 2
  for (int kc = 0; kc < KTOT / 32; ++kc) {
    int tap = kc / CPT;
    int cib = (kc % CPT) * 32;
    int dy = tap / KS, dx = tap % KS;
    v16h a = afrag(wrow + tap * CIN + cib, g);
    v16h b = bfrag(&smem[dy][n + dx][cib], g);
    acc = wmma16(a, b, acc);
  }
  int x = x0 + n;
  if (x < WL) {
#pragma unroll
    for (int r = 0; r < 8; ++r) {
      int co = co0 + r + g * 8;
      float v = acc[r] + bias[co];
      size_t oi = ((size_t)co * HL + y) * WL + x;
      if (addSrc) v += addSrc[oi];
      if (reluOut) v = v > 0.f ? v : 0.f;
      out[oi] = v;
    }
  }
}

// ---------------- GroupNorm(16 groups over C=128) + ReLU ---------------------------
__global__ void __launch_bounds__(256)
gn_relu(const float* __restrict__ in, const float* __restrict__ gam,
        const float* __restrict__ bet, float* __restrict__ out, int H, int W) {
  __shared__ float s1[256], s2[256];
  const int grp = blockIdx.x;
  const int c0 = grp * 8;
  const int hw = H * W;
  const int n = 8 * hw;
  const float* basep = in + (size_t)c0 * hw;
  float a = 0.f, b = 0.f;
  for (int i = threadIdx.x; i < n; i += 256) {
    float v = basep[i];
    a += v;
    b += v * v;
  }
  s1[threadIdx.x] = a;
  s2[threadIdx.x] = b;
  __syncthreads();
  for (int o = 128; o > 0; o >>= 1) {
    if (threadIdx.x < o) {
      s1[threadIdx.x] += s1[threadIdx.x + o];
      s2[threadIdx.x] += s2[threadIdx.x + o];
    }
    __syncthreads();
  }
  float m = s1[0] / n;
  float var = s2[0] / n - m * m;
  if (var < 0.f) var = 0.f;
  float rstd = rsqrtf(var + 1e-6f);
  float* op = out + (size_t)c0 * hw;
  for (int i = threadIdx.x; i < n; i += 256) {
    int c = c0 + i / hw;
    float v = (basep[i] - m) * rstd * gam[c] + bet[c];
    op[i] = v > 0.f ? v : 0.f;
  }
}

// ---------------- x0 = concat(feat_up, up4(fl)) gathered per token -----------------
__global__ void build_x0(const float* __restrict__ featUp, const float* __restrict__ fl,
                         float* __restrict__ x, int Hu, int Wu) {
  int i = blockIdx.x * 256 + threadIdx.x;
  int tot = Hu * Wu * 192;
  if (i >= tot) return;
  int c = i % 192;
  int tok = i / 192;
  int y = tok / Wu, xx = tok % Wu;
  float v;
  if (c < 128)
    v = featUp[((size_t)c * Hu + y) * Wu + xx];
  else
    v = fl[(((size_t)(c - 128)) * (Hu / 4) + (y >> 2)) * (Wu / 4) + (xx >> 2)];
  x[i] = v;
}

// ---------------- K/V prep: LN_affine(ctx_src) @ Wk^T/Wv^T over 1601 rows ----------
__global__ void __launch_bounds__(256)
kv_prep(const float* __restrict__ feat, const float* __restrict__ fl,
        const float* __restrict__ ctxg, const float* __restrict__ ctxb,
        const h16* __restrict__ Wk, const float* __restrict__ kb,
        const h16* __restrict__ Wv, const float* __restrict__ vb,
        float* __restrict__ Kdst, float* __restrict__ Vdst) {
  __shared__ float raw[16][192];
  __shared__ h16 sa[16][200];
  __shared__ float mu[16], rs[16];
  const int tid = threadIdx.x;
  const int s0 = blockIdx.x * 16;
  for (int i = tid; i < 16 * 192; i += 256) {
    int c = i % 192, r = i / 192;
    int s = s0 + r;
    float v = 0.f;
    if (s < 1600) {
      int ly = s / 40, lx = s % 40;
      v = (c < 128) ? feat[((size_t)c * 40 + ly) * 40 + lx]
                    : fl[((size_t)(c - 128) * 40 + ly) * 40 + lx];
    }
    raw[r][c] = v;
  }
  __syncthreads();
  if (tid < 16) {
    float a = 0.f, b = 0.f;
    for (int c = 0; c < 192; ++c) {
      float v = raw[tid][c];
      a += v;
      b += v * v;
    }
    float m = a / 192.f;
    float var = b / 192.f - m * m;
    if (var < 0.f) var = 0.f;
    mu[tid] = m;
    rs[tid] = rsqrtf(var + 1e-5f);  // torch-default eps for the affine ctx LN
  }
  __syncthreads();
  for (int i = tid; i < 16 * 192; i += 256) {
    int c = i % 192, r = i / 192;
    sa[r][c] = (h16)((raw[r][c] - mu[r]) * rs[r] * ctxg[c] + ctxb[c]);
  }
  __syncthreads();
  const int w = tid >> 5, lane = tid & 31, n = lane & 15, g = lane >> 4;
  for (int task = w; task < 24; task += 8) {
    int ct = task % 12;
    const h16* W = (task < 12) ? Wk : Wv;
    const float* bias = (task < 12) ? kb : vb;
    float* dst = (task < 12) ? Kdst : Vdst;
    int col = ct * 16 + n;
    v8f acc = {};
#pragma unroll
    for (int kc = 0; kc < 6; ++kc) {
      v16h a = afrag(&sa[n][kc * 32], g);
      v16h b = bfrag(W + (size_t)col * 192 + kc * 32, g);
      acc = wmma16(a, b, acc);
    }
#pragma unroll
    for (int r = 0; r < 8; ++r) {
      int s = s0 + r + g * 8;
      if (s <= 1600) dst[(size_t)s * 192 + col] = acc[r] + bias[col];
    }
  }
}

__device__ __forceinline__ float gelu_tanh(float v) {
  return 0.5f * v * (1.0f + tanhf(0.7978845608028654f * (v + 0.044715f * v * v * v)));
}

// ---------------- fused cross-attention block over a 16-token tile -----------------
__global__ void __launch_bounds__(256)
attn_block(float* __restrict__ x, const float* __restrict__ Ks,
           const float* __restrict__ Vs, const h16* __restrict__ Wq,
           const float* __restrict__ bq, const h16* __restrict__ Wo,
           const float* __restrict__ bo, const h16* __restrict__ W1,
           const float* __restrict__ b1, const h16* __restrict__ W2,
           const float* __restrict__ b2) {
  __shared__ float sx[16][192];   // running hidden state (f32)
  __shared__ h16 sa[16][200];     // staged f16 GEMM-A operand (xn / o / xn2)
  __shared__ float sq[16][192];   // q
  __shared__ h16 sh[16][776];     // gelu(fc1) staged for fc2
  __shared__ float mu[16], rs[16];
  const int tid = threadIdx.x;
  const size_t n0 = (size_t)blockIdx.x * 16;

  for (int i = tid; i < 16 * 192; i += 256) sx[i / 192][i % 192] = x[n0 * 192 + i];
  __syncthreads();
  // LN (no affine, eps 1e-6)
  if (tid < 16) {
    float a = 0.f, b = 0.f;
    for (int c = 0; c < 192; ++c) {
      float v = sx[tid][c];
      a += v;
      b += v * v;
    }
    float m = a / 192.f;
    float var = b / 192.f - m * m;
    if (var < 0.f) var = 0.f;
    mu[tid] = m;
    rs[tid] = rsqrtf(var + 1e-6f);
  }
  __syncthreads();
  for (int i = tid; i < 16 * 192; i += 256) {
    int r = i / 192, c = i % 192;
    sa[r][c] = (h16)((sx[r][c] - mu[r]) * rs[r]);
  }
  __syncthreads();

  const int w = tid >> 5, lane = tid & 31, n = lane & 15, g = lane >> 4;
  // q = xn @ Wq^T + bq
  for (int ct = w; ct < 12; ct += 8) {
    int col = ct * 16 + n;
    v8f acc = {};
#pragma unroll
    for (int kc = 0; kc < 6; ++kc) {
      v16h a = afrag(&sa[n][kc * 32], g);
      v16h b = bfrag(Wq + (size_t)col * 192 + kc * 32, g);
      acc = wmma16(a, b, acc);
    }
#pragma unroll
    for (int r = 0; r < 8; ++r) sq[r + g * 8][col] = acc[r] + bq[col];
  }
  __syncthreads();

  // 9-key ALiBi attention; one thread per (token, head); writes o (f16) into sa
  if (tid < 64) {
    int r = tid >> 2, h = tid & 3;
    int tok = (int)(n0 + r);
    int yy = tok / 160, xx = tok % 160;
    int ly = yy >> 2, lx = xx >> 2;
    const float invs = 0.14433756729740643f;  // 1/sqrt(48)
    float slope = (h & 1) ? (1.0f / 24.0f) : 0.20412414523193154f;
    bool useY = (h >= 2);
    float sc[9];
    int id[9];
    float mx = -1e30f;
#pragma unroll
    for (int t = 0; t < 9; ++t) {
      int ti = t / 3, tj = t % 3;
      int ky = ly - 1 + ti, kx = lx - 1 + tj;
      int kid = ((unsigned)ky < 40u && (unsigned)kx < 40u) ? ky * 40 + kx : 1600;
      id[t] = kid;
      const float* kp = Ks + (size_t)kid * 192 + h * 48;
      float d = 0.f;
      for (int c = 0; c < 48; ++c) d += sq[r][h * 48 + c] * kp[c];
      float coord = -fabsf((float)(useY ? ti : tj) - 1.0f);
      d = d * invs + slope * coord;
      sc[t] = d;
      mx = fmaxf(mx, d);
    }
    float sum = 0.f;
#pragma unroll
    for (int t = 0; t < 9; ++t) {
      sc[t] = __expf(sc[t] - mx);
      sum += sc[t];
    }
    float inv = 1.0f / sum;
    for (int c = 0; c < 48; ++c) {
      float o = 0.f;
#pragma unroll
      for (int t = 0; t < 9; ++t) o += sc[t] * Vs[(size_t)id[t] * 192 + h * 48 + c];
      sa[r][h * 48 + c] = (h16)(o * inv);
    }
  }
  __syncthreads();

  // x += o @ Wo^T + bo
  for (int ct = w; ct < 12; ct += 8) {
    int col = ct * 16 + n;
    v8f acc = {};
#pragma unroll
    for (int kc = 0; kc < 6; ++kc) {
      v16h a = afrag(&sa[n][kc * 32], g);
      v16h b = bfrag(Wo + (size_t)col * 192 + kc * 32, g);
      acc = wmma16(a, b, acc);
    }
#pragma unroll
    for (int r = 0; r < 8; ++r) sx[r + g * 8][col] += acc[r] + bo[col];
  }
  __syncthreads();

  // LN (no affine) -> sa
  if (tid < 16) {
    float a = 0.f, b = 0.f;
    for (int c = 0; c < 192; ++c) {
      float v = sx[tid][c];
      a += v;
      b += v * v;
    }
    float m = a / 192.f;
    float var = b / 192.f - m * m;
    if (var < 0.f) var = 0.f;
    mu[tid] = m;
    rs[tid] = rsqrtf(var + 1e-6f);
  }
  __syncthreads();
  for (int i = tid; i < 16 * 192; i += 256) {
    int r = i / 192, c = i % 192;
    sa[r][c] = (h16)((sx[r][c] - mu[r]) * rs[r]);
  }
  __syncthreads();

  // fc1 (192 -> 768) + GELU -> sh
  for (int ct = w; ct < 48; ct += 8) {
    int col = ct * 16 + n;
    v8f acc = {};
#pragma unroll
    for (int kc = 0; kc < 6; ++kc) {
      v16h a = afrag(&sa[n][kc * 32], g);
      v16h b = bfrag(W1 + (size_t)col * 192 + kc * 32, g);
      acc = wmma16(a, b, acc);
    }
#pragma unroll
    for (int r = 0; r < 8; ++r)
      sh[r + g * 8][col] = (h16)gelu_tanh(acc[r] + b1[col]);
  }
  __syncthreads();

  // x += fc2(h) (768 -> 192)
  for (int ct = w; ct < 12; ct += 8) {
    int col = ct * 16 + n;
    v8f acc = {};
#pragma unroll 4
    for (int kc = 0; kc < 24; ++kc) {
      v16h a = afrag(&sh[n][kc * 32], g);
      v16h b = bfrag(W2 + (size_t)col * 768 + kc * 32, g);
      acc = wmma16(a, b, acc);
    }
#pragma unroll
    for (int r = 0; r < 8; ++r) sx[r + g * 8][col] += acc[r] + b2[col];
  }
  __syncthreads();

  for (int i = tid; i < 16 * 192; i += 256) x[n0 * 192 + i] = sx[i / 192][i % 192];
}

// ---------------- final 192 -> 9 projection + softmax, NHWC -> NCHW ----------------
__global__ void out_softmax(const float* __restrict__ x, const float* __restrict__ ow,
                            const float* __restrict__ ob, float* __restrict__ out) {
  int tok = blockIdx.x * 256 + threadIdx.x;
  if (tok >= 25600) return;
  const float* xr = x + (size_t)tok * 192;
  float lg[9];
  float mx = -1e30f;
#pragma unroll
  for (int t = 0; t < 9; ++t) {
    float d = ob[t];
    for (int c = 0; c < 192; ++c) d += xr[c] * ow[t * 192 + c];
    lg[t] = d;
    mx = fmaxf(mx, d);
  }
  float s = 0.f;
#pragma unroll
  for (int t = 0; t < 9; ++t) {
    lg[t] = __expf(lg[t] - mx);
    s += lg[t];
  }
  float inv = 1.0f / s;
  int y = tok / 160, xx = tok % 160;
#pragma unroll
  for (int t = 0; t < 9; ++t) out[((size_t)t * 160 + y) * 160 + xx] = lg[t] * inv;
}

// ===================================================================================
extern "C" void kernel_launch(void* const* d_in, const int* in_sizes, int n_in,
                              void* d_out, int out_size, void* d_ws, size_t ws_size,
                              hipStream_t stream) {
  (void)in_sizes; (void)n_in; (void)out_size;
  auto F = [&](int i) { return (const float*)d_in[i]; };
  // input index map (setup_inputs dict order, params flattened in insertion order)
  const float* feat = F(0);  // [1,128,40,40]
  const float* flow = F(1);  // [1,32,3,40,40] == [96,40,40]
  const int CONV_IN_W = 2, CONV_IN_B = 3, N1_G = 4, N1_B = 5;
  const int RB0 = 6;   // gn1_g,gn1_b,c1_w,c1_b,gn2_g,gn2_b,c2_w,c2_b,up_w,up_b
  const int RB1 = 16;
  const int N2_G = 26, N2_B = 27, COUT_W = 28, COUT_B = 29;
  const int F1_W = 30, F1_B = 31, F2_W = 32, F2_B = 33;
  const int BLK0 = 34, BLK1 = 48;  // ctx_g,ctx_b,q_w,q_b,k_w,k_b,v_w,v_b,o_w,o_b,fc1_w,fc1_b,fc2_w,fc2_b
  const int OUT_W = 62, OUT_B = 63;

  // ---- workspace carve-out ----
  char* base = (char*)d_ws;
  size_t off = 0;
  auto alloc = [&](size_t bytes) -> void* {
    off = (off + 255) & ~(size_t)255;
    void* p = base + off;
    off += bytes;
    return p;
  };
  const size_t HW160 = 160 * 160;
  float* t0 = (float*)alloc(128 * HW160 * 4);
  float* t1 = (float*)alloc(128 * HW160 * 4);
  float* t2 = (float*)alloc(128 * HW160 * 4);
  float* fb1 = (float*)alloc(128 * 1600 * 4);
  float* flb = (float*)alloc(64 * 1600 * 4);
  float* xb = (float*)alloc((size_t)25600 * 192 * 4);
  float* Kb[2], *Vb[2];
  for (int i = 0; i < 2; ++i) {
    Kb[i] = (float*)alloc((size_t)1616 * 192 * 4);
    Vb[i] = (float*)alloc((size_t)1616 * 192 * 4);
  }
  h16* pcIn = (h16*)alloc(147456 * 2);
  h16* pc1[2], *pc2[2], *pup[2];
  for (int i = 0; i < 2; ++i) {
    pc1[i] = (h16*)alloc(147456 * 2);
    pc2[i] = (h16*)alloc(147456 * 2);
    pup[i] = (h16*)alloc(147456 * 2);
  }
  h16* pcOut = (h16*)alloc(147456 * 2);
  h16* pf1 = (h16*)alloc((size_t)602112 * 2);
  h16* pf2 = (h16*)alloc(73728 * 2);
  h16* pq[2], *pk[2], *pv[2], *po[2], *pm1[2], *pm2[2];
  for (int i = 0; i < 2; ++i) {
    pq[i] = (h16*)alloc(36864 * 2);
    pk[i] = (h16*)alloc(36864 * 2);
    pv[i] = (h16*)alloc(36864 * 2);
    po[i] = (h16*)alloc(36864 * 2);
    pm1[i] = (h16*)alloc(147456 * 2);
    pm2[i] = (h16*)alloc(147456 * 2);
  }
  if (off > ws_size) return;  // insufficient scratch

  auto packc = [&](const float* w, h16* d, int CO, int CI, int KS) {
    int n = CO * CI * KS * KS;
    pack_conv<<<(n + 255) / 256, 256, 0, stream>>>(w, d, CO, CI, KS);
  };
  auto packm = [&](const float* w, h16* d, int n) {
    pack_mat<<<(n + 255) / 256, 256, 0, stream>>>(w, d, n);
  };
  // ---- pack all weights to f16 ----
  packc(F(CONV_IN_W), pcIn, 128, 128, 3);
  packc(F(RB0 + 2), pc1[0], 128, 128, 3);
  packc(F(RB0 + 6), pc2[0], 128, 128, 3);
  packc(F(RB0 + 8), pup[0], 128, 128, 3);
  packc(F(RB1 + 2), pc1[1], 128, 128, 3);
  packc(F(RB1 + 6), pc2[1], 128, 128, 3);
  packc(F(RB1 + 8), pup[1], 128, 128, 3);
  packc(F(COUT_W), pcOut, 128, 128, 3);
  packc(F(F1_W), pf1, 128, 96, 7);
  packc(F(F2_W), pf2, 64, 128, 3);
  const int BB[2] = {BLK0, BLK1};
  for (int i = 0; i < 2; ++i) {
    packm(F(BB[i] + 2), pq[i], 36864);
    packm(F(BB[i] + 4), pk[i], 36864);
    packm(F(BB[i] + 6), pv[i], 36864);
    packm(F(BB[i] + 8), po[i], 36864);
    packm(F(BB[i] + 10), pm1[i], 147456);
    packm(F(BB[i] + 12), pm2[i], 147456);
  }

  dim3 blk(256);
  dim3 g40(3, 40), g80(5, 80), g160(10, 160);

  // ---- decoder upsampler ----
  conv_wmma<3, 128, 128, false><<<g40, blk, 0, stream>>>(feat, 40, 40, pcIn, F(CONV_IN_B), nullptr, 0, t0);
  gn_relu<<<16, blk, 0, stream>>>(t0, F(N1_G), F(N1_B), t1, 40, 40);
  // residual block 0 @40, then up->80
  gn_relu<<<16, blk, 0, stream>>>(t1, F(RB0 + 0), F(RB0 + 1), t2, 40, 40);
  conv_wmma<3, 128, 128, false><<<g40, blk, 0, stream>>>(t2, 40, 40, pc1[0], F(RB0 + 3), nullptr, 0, t0);
  gn_relu<<<16, blk, 0, stream>>>(t0, F(RB0 + 4), F(RB0 + 5), t2, 40, 40);
  conv_wmma<3, 128, 128, false><<<g40, blk, 0, stream>>>(t2, 40, 40, pc2[0], F(RB0 + 7), t1, 0, t0);
  conv_wmma<3, 128, 128, true><<<g80, blk, 0, stream>>>(t0, 40, 40, pup[0], F(RB0 + 9), nullptr, 0, t1);
  // residual block 1 @80, then up->160
  gn_relu<<<16, blk, 0, stream>>>(t1, F(RB1 + 0), F(RB1 + 1), t2, 80, 80);
  conv_wmma<3, 128, 128, false><<<g80, blk, 0, stream>>>(t2, 80, 80, pc1[1], F(RB1 + 3), nullptr, 0, t0);
  gn_relu<<<16, blk, 0, stream>>>(t0, F(RB1 + 4), F(RB1 + 5), t2, 80, 80);
  conv_wmma<3, 128, 128, false><<<g80, blk, 0, stream>>>(t2, 80, 80, pc2[1], F(RB1 + 7), t1, 0, t0);
  conv_wmma<3, 128, 128, true><<<g160, blk, 0, stream>>>(t0, 80, 80, pup[1], F(RB1 + 9), nullptr, 0, t1);
  gn_relu<<<16, blk, 0, stream>>>(t1, F(N2_G), F(N2_B), t2, 160, 160);
  conv_wmma<3, 128, 128, false><<<g160, blk, 0, stream>>>(t2, 160, 160, pcOut, F(COUT_B), nullptr, 0, t0);
  // t0 = feat_up [128,160,160]

  // ---- flow path ----
  conv_wmma<7, 96, 128, false><<<g40, blk, 0, stream>>>(flow, 40, 40, pf1, F(F1_B), nullptr, 1, fb1);
  conv_wmma<3, 128, 64, false><<<g40, blk, 0, stream>>>(fb1, 40, 40, pf2, F(F2_B), nullptr, 1, flb);

  // ---- token assembly + K/V prep (1601 distinct ctx vectors per block) ----
  build_x0<<<(25600 * 192 + 255) / 256, blk, 0, stream>>>(t0, flb, xb, 160, 160);
  for (int i = 0; i < 2; ++i) {
    kv_prep<<<101, blk, 0, stream>>>(feat, flb, F(BB[i] + 0), F(BB[i] + 1), pk[i],
                                     F(BB[i] + 5), pv[i], F(BB[i] + 7), Kb[i], Vb[i]);
  }
  // ---- two fused cross-attention blocks ----
  for (int i = 0; i < 2; ++i) {
    attn_block<<<1600, blk, 0, stream>>>(xb, Kb[i], Vb[i], pq[i], F(BB[i] + 3), po[i],
                                         F(BB[i] + 9), pm1[i], F(BB[i] + 11), pm2[i],
                                         F(BB[i] + 13));
  }
  // ---- final projection + softmax ----
  out_softmax<<<100, blk, 0, stream>>>(xb, F(OUT_W), F(OUT_B), (float*)d_out);
}
